// TopologyLayer_25898652795463
// MI455X (gfx1250) — matrix operand order repaired and checked by reading, not compiled, not run on hardware
//
#include <hip/hip_runtime.h>
#include <stdint.h>

// ---------------------------------------------------------------------------
// MI455X (gfx1250) implementation.
// Roofline: 13.5 GFLOP total, ~137 MB HBM traffic -> ~5.9us memory floor.
// bf16 WMMA (v_wmma_f32_16x16x32_bf16) keeps GEMM compute (~2-3us) under the
// memory floor; f32 WMMA (16x16x4) would be 8x slower per instruction and
// compute-bound. So: one-time f32->bf16 conversion of x and W_out, LDS tile
// staging, fused epilogues. B k-slices of the big GEMM are staged by the
// Tensor Data Mover (tensor_load_to_lds + s_wait_tensorcnt) when available:
// a 2D D# with LDS padding reproduces the padded [256][40] LDS layout with
// zero VALU/VGPR cost, double-buffered against WMMA.
// ---------------------------------------------------------------------------

typedef unsigned short u16;
typedef __attribute__((ext_vector_type(16))) __bf16 bf16x16;
typedef __attribute__((ext_vector_type(8)))  float  f32x8;
typedef __attribute__((ext_vector_type(4)))  unsigned int u32x4;
typedef __attribute__((ext_vector_type(8)))  int i32x8;
typedef __attribute__((ext_vector_type(4)))  int i32x4;

#if defined(__AMDGCN__) && __has_builtin(__builtin_amdgcn_tensor_load_to_lds) && \
    __has_builtin(__builtin_amdgcn_s_wait_tensorcnt)
#define USE_TDM 1
#else
#define USE_TDM 0
#endif

union Frag {
  bf16x16 bf;
  uint4   q[2];
};

__device__ __forceinline__ u16 f2bf(float f) {
  union { float f; uint32_t u; } v; v.f = f;
  uint32_t r = v.u + 0x7FFFu + ((v.u >> 16) & 1u);   // round-nearest-even
  return (u16)(r >> 16);
}
__device__ __forceinline__ uint32_t pack2(float lo, float hi) {
  return (uint32_t)f2bf(lo) | ((uint32_t)f2bf(hi) << 16);
}
// Load a 16x32-bf16 A/B fragment's per-lane 2x16B chunks from LDS.
__device__ __forceinline__ bf16x16 ld_frag2(const u16* p0, const u16* p1) {
  Frag f;
  f.q[0] = *(const uint4*)p0;
  f.q[1] = *(const uint4*)p1;
  return f.bf;
}
__device__ __forceinline__ float4 max4(float4 a, float4 b) {
  return make_float4(fmaxf(a.x, b.x), fmaxf(a.y, b.y),
                     fmaxf(a.z, b.z), fmaxf(a.w, b.w));
}

// The 4 coordinate transforms (tri, gau, lin, rh), D=3 each -> 12 features.
__device__ __forceinline__ void coord12(float x0, float x1,
    const float* __restrict__ t_tri, const float* __restrict__ t_gau,
    float sig, const float* __restrict__ W_lin, const float* __restrict__ b_lin,
    const float* __restrict__ c_rh, float r_rh, float* __restrict__ out) {
  const float inv2s2 = 1.0f / (2.0f * sig * sig);
  const float ar = fabsf(r_rh);
#pragma unroll
  for (int d = 0; d < 3; ++d) {
    float tri = fmaxf(x1 - fabsf(t_tri[d] - x0), 0.0f);
    float dx = x0 - t_gau[2 * d + 0], dy = x1 - t_gau[2 * d + 1];
    float gau = __expf(-(dx * dx + dy * dy) * inv2s2);
    float lin = x0 * W_lin[d] + x1 * W_lin[3 + d] + b_lin[d];
    float l1 = fabsf(x0 - c_rh[d]) + fabsf(x1 - c_rh[d]);
    float rh = 1.0f / (1.0f + l1) - 1.0f / (1.0f + fabsf(ar - l1));
    out[d] = tri; out[3 + d] = gau; out[6 + d] = lin; out[9 + d] = rh;
  }
}

constexpr int NG = 128, NPG = 400, EPG = 6400;
constexpr int NN = NG * NPG;            // 51200 nodes
constexpr int NE = NG * EPG;            // 819200 edges
constexpr int FIN = 256, HIDN = 64, KP = 8;
constexpr int FCAT = FIN + 2 * 96;      // 448

// ----------------------- kernel: W_out -> bf16, transposed [256][448] -------
__global__ void wt_kernel(const float* __restrict__ W_out, u16* __restrict__ WT) {
  int i = blockIdx.x * 256 + threadIdx.x;          // 448*256 elements
  if (i >= FCAT * FIN) return;
  int k = i >> 8, n = i & 255;                     // W_out row-major [448][256]
  WT[(size_t)n * FCAT + k] = f2bf(W_out[i]);
}

// ----------------------- kernel: fv = relu(relu(x@W1+b1)@W2+b2) ------------
// 400 blocks x 256 threads (8 waves); each block: 128 rows. Also emits x_bf16.
// LDS: sX[128][264] + sW1t[64][264] + sW2t[16][72] + sH[8][16][72]  (u16)
constexpr int K0_SX  = 0;
constexpr int K0_SW1 = K0_SX  + 128 * 264 * 2;     // 67584
constexpr int K0_SW2 = K0_SW1 + 64 * 264 * 2;      // +33792 = 101376
constexpr int K0_SH  = K0_SW2 + 16 * 72 * 2;       // +2304  = 103680
constexpr int K0_LDS = K0_SH  + 8 * 16 * 72 * 2;   // +18432 = 122112

__global__ void fv_kernel(const float* __restrict__ x,
                          const float* __restrict__ W1, const float* __restrict__ b1,
                          const float* __restrict__ W2, const float* __restrict__ b2,
                          float* __restrict__ fv, u16* __restrict__ xb) {
  extern __shared__ char smem[];
  u16* sX   = (u16*)(smem + K0_SX);
  u16* sW1t = (u16*)(smem + K0_SW1);
  u16* sW2t = (u16*)(smem + K0_SW2);
  u16* sH   = (u16*)(smem + K0_SH);

  const int t = threadIdx.x;
  const int row0 = blockIdx.x * 128;

  for (int i = t; i < 16 * 72; i += 256) sW2t[i] = 0;   // zero-pad W2 cols 8..15
  __syncthreads();

  // stage x tile (f32 -> bf16 in LDS) and write back x_bf16 to global ws
#pragma unroll
  for (int i = 0; i < 32; ++i) {
    int u = t + i * 256;                 // float4 index, 64 per row
    int r = u >> 6, c4 = u & 63;
    float4 v = *(const float4*)(x + (size_t)(row0 + r) * FIN + c4 * 4);
    uint2 pk; pk.x = pack2(v.x, v.y); pk.y = pack2(v.z, v.w);
    *(uint2*)&sX[r * 264 + c4 * 4] = pk;
    *(uint2*)(xb + (size_t)(row0 + r) * FIN + c4 * 4) = pk;
  }
  // stage W1 transposed [64][264]
#pragma unroll
  for (int i = 0; i < 64; ++i) {
    int idx = t + i * 256;               // [256][64] row-major
    int k = idx >> 6, n = idx & 63;
    sW1t[n * 264 + k] = f2bf(W1[idx]);
  }
  // stage W2 transposed [16][72] (rows 8..15 already zero)
  for (int i = t; i < HIDN * KP; i += 256) {
    int k = i >> 3, n = i & 7;
    sW2t[n * 72 + k] = f2bf(W2[i]);
  }
  __syncthreads();

  const int w = t >> 5, lane = t & 31;
  const int lrow = lane & 15, lhi = lane >> 4;

  f32x8 zero = {};
  f32x8 acc[4];
#pragma unroll
  for (int j = 0; j < 4; ++j) acc[j] = zero;

  const u16* arow = sX + (w * 16 + lrow) * 264;
#pragma unroll
  for (int ks = 0; ks < 8; ++ks) {
    int k0 = ks * 32;
    int kA = k0 + 16 * lhi;
    bf16x16 a = ld_frag2(arow + kA, arow + kA + 8);
#pragma unroll
    for (int jn = 0; jn < 4; ++jn) {
      const u16* bp = sW1t + (jn * 16 + lrow) * 264 + k0 + 16 * lhi;
      bf16x16 b = ld_frag2(bp, bp + 8);
      acc[jn] = __builtin_amdgcn_wmma_f32_16x16x32_bf16(
          false, a, false, b, (short)0, acc[jn], false, false);
    }
  }

  // h = relu(acc + b1) -> LDS (per-wave private region), bf16 row-major [16][72]
  u16* hrow = sH + w * (16 * 72);
#pragma unroll
  for (int jn = 0; jn < 4; ++jn) {
    int col = jn * 16 + lrow;
    float bb = b1[col];
#pragma unroll
    for (int r = 0; r < 8; ++r) {
      int row = r + 8 * lhi;
      hrow[row * 72 + col] = f2bf(fmaxf(acc[jn][r] + bb, 0.0f));
    }
  }

  // second GEMM: h[16x64] @ W2t -> 16x16 (cols 0..7 valid)
  f32x8 c2 = zero;
#pragma unroll
  for (int ks = 0; ks < 2; ++ks) {
    int k0 = ks * 32;
    const u16* ar = hrow + lrow * 72;
    bf16x16 a = ld_frag2(ar + k0 + 16 * lhi, ar + k0 + 16 * lhi + 8);
    const u16* bp = sW2t + lrow * 72 + k0 + 16 * lhi;
    bf16x16 b = ld_frag2(bp, bp + 8);
    c2 = __builtin_amdgcn_wmma_f32_16x16x32_bf16(
        false, a, false, b, (short)0, c2, false, false);
  }
  if (lrow < 8) {
    float bb = b2[lrow];
#pragma unroll
    for (int r = 0; r < 8; ++r) {
      int row = w * 16 + r + 8 * lhi;
      fv[(size_t)(row0 + row) * KP + lrow] = fmaxf(c2[r] + bb, 0.0f);
    }
  }
}

// ----------------------- kernel: per-graph edge max -> unpaired[128][8] -----
__global__ void edge_max_kernel(const int* __restrict__ ei,
                                const float* __restrict__ fv,
                                float* __restrict__ unpaired) {
  __shared__ float4 red[256][2];
  const int g = blockIdx.x, t = threadIdx.x;
  const int* ei0 = ei;
  const int* ei1 = ei + NE;
  float4 m0 = make_float4(-3.0e38f, -3.0e38f, -3.0e38f, -3.0e38f);
  float4 m1 = m0;
  for (int e = g * EPG + t; e < (g + 1) * EPG; e += 256) {
    int a = ei0[e], b = ei1[e];
    const float4* pa = (const float4*)(fv + (size_t)a * KP);
    const float4* pb = (const float4*)(fv + (size_t)b * KP);
    m0 = max4(m0, max4(pa[0], pb[0]));
    m1 = max4(m1, max4(pa[1], pb[1]));
  }
  red[t][0] = m0; red[t][1] = m1;
  __syncthreads();
  for (int s = 128; s > 0; s >>= 1) {
    if (t < s) {
      red[t][0] = max4(red[t][0], red[t + s][0]);
      red[t][1] = max4(red[t][1], red[t + s][1]);
    }
    __syncthreads();
  }
  if (t == 0) {
    *(float4*)(unpaired + (size_t)g * KP)     = red[0][0];
    *(float4*)(unpaired + (size_t)g * KP + 4) = red[0][1];
  }
}

// ----------------------- kernel: g1 (masked coord feats of (u,0)) -> bf16 ---
__global__ void g1_kernel(const float* __restrict__ unpaired, u16* __restrict__ g1b,
    const float* t_tri1, const float* t_gau1, const float* sig1,
    const float* W_lin1, const float* b_lin1, const float* c_rh1,
    const float* r_rh1) {
  int g = threadIdx.x;
  if (g >= NG) return;
  float u[KP]; bool any = false;
#pragma unroll
  for (int k = 0; k < KP; ++k) { u[k] = unpaired[g * KP + k]; any = any || (u[k] != 0.0f); }
  const float sg = sig1[0], rr = r_rh1[0];
  float f[12];
#pragma unroll
  for (int k = 0; k < KP; ++k) {
    coord12(u[k], 0.0f, t_tri1, t_gau1, sg, W_lin1, b_lin1, c_rh1, rr, f);
#pragma unroll
    for (int j = 0; j < 12; ++j)
      g1b[g * 96 + k * 12 + j] = any ? f2bf(f[j]) : (u16)0;
  }
}

// ----------------------- kernel: out = relu([x|act0|g1]@W_out + b_out) -----
// 400 blocks x 256 threads (8 waves). Block tile: 128 rows x 256 cols.
// A in LDS: [128][456] bf16 (x | act0-from-fv | g1[batch]).
// B double-buffered k-slices: [2][256][40] bf16 (column-major per n),
// staged by the Tensor Data Mover when available.
constexpr int K3_SA  = 0;
constexpr int K3_SB  = K3_SA + 128 * 456 * 2;      // 116736
constexpr int K3_SBUF = 256 * 40 * 2;              // 20480 per buffer
constexpr int K3_LDS = K3_SB + 2 * K3_SBUF;        // 157696

#if USE_TDM
// Issue one TDM 2D tile load: WT[0..255][ks*32 .. ks*32+31] (bf16, row stride
// 448) -> LDS rows of 64B with 16B padding => sB[buf][n][40] layout.
__device__ __forceinline__ void tdm_load_bslice(const u16* WT, int ks, int buf) {
  uint64_t gaddr = (uint64_t)(uintptr_t)(WT + (size_t)ks * 32);
  uint32_t lds   = (uint32_t)(K3_SB + buf * K3_SBUF);
  u32x4 g0;
  g0.x = 1u;                                         // count=1 (valid, user)
  g0.y = lds;                                        // lds_addr (bytes)
  g0.z = (uint32_t)gaddr;                            // global_addr[31:0]
  g0.w = (uint32_t)((gaddr >> 32) & 0x1FFFFFFull)    // global_addr[56:32]
         | 0x80000000u;                              // type=2 ("image")
  i32x8 g1;
  g1[0] = (int)((1u << 16)        // data_size = 2 bytes
              | (1u << 20)        // pad_enable
              | (3u << 22)        // pad_interval: 16 dwords (64B)
              | (3u << 25));      // pad_amount:   4 dwords (16B)
  g1[1] = (int)(448u << 16);      // tensor_dim0 = 448  (bits 79:48)
  g1[2] = (int)(256u << 16);      // tensor_dim1 = 256  (bits 111:80)
  g1[3] = (int)(32u  << 16);      // tile_dim0   = 32   (bits 127:112)
  g1[4] = 256;                    // tile_dim1   = 256  (bits 143:128)
  g1[5] = 448;                    // tensor_dim0_stride = 448 (bits 207:160)
  g1[6] = 0;
  g1[7] = 0;
  i32x4 z4 = {0, 0, 0, 0};
  i32x8 z8 = {0, 0, 0, 0, 0, 0, 0, 0};
  // 6-arg toolchain flavor: (g0, g1, g2, g3, extra group, cpol)
  __builtin_amdgcn_tensor_load_to_lds(g0, g1, z4, z4, z8, 0);
}
#endif

__global__ void out_gemm_kernel(const u16* __restrict__ xb,
    const float* __restrict__ fv, const u16* __restrict__ g1b,
    const u16* __restrict__ WT, const float* __restrict__ b_out,
    const float* t_tri0, const float* t_gau0, const float* sig0,
    const float* W_lin0, const float* b_lin0, const float* c_rh0,
    const float* r_rh0, float* __restrict__ out) {
  extern __shared__ char smem[];
  u16* sA = (u16*)(smem + K3_SA);
  u16* sB = (u16*)(smem + K3_SB);

  const int t = threadIdx.x;
  const int row0 = blockIdx.x * 128;
  const int w = t >> 5, lane = t & 31;
  const int lrow = lane & 15, lhi = lane >> 4;

#if USE_TDM
  if (w == 0) tdm_load_bslice(WT, 0, 0);   // DMA slice 0 while we stage A
#endif

  // ---- stage A: x_bf16 part (cols 0..255) ----
#pragma unroll
  for (int i = 0; i < 16; ++i) {
    int u = t + i * 256;                 // uint4 (8 bf16) index, 32 per row
    int r = u >> 5, c8 = (u & 31) << 3;
    uint4 q = *(const uint4*)(xb + (size_t)(row0 + r) * FIN + c8);
    *(uint4*)&sA[r * 456 + c8] = q;
  }
  // ---- stage A: act0 (cols 256..351) + g1[batch] (cols 352..399) ----
  if (t < 128) {
    const int r = t, grow = row0 + r;
    const float sg = sig0[0], rr = r_rh0[0];
    float f[12];
#pragma unroll
    for (int k = 0; k < KP; ++k) {
      float v = fv[(size_t)grow * KP + k];     // pers0 pair = (v, v)
      coord12(v, v, t_tri0, t_gau0, sg, W_lin0, b_lin0, c_rh0, rr, f);
#pragma unroll
      for (int j = 0; j < 12; ++j)
        sA[r * 456 + 256 + k * 12 + j] = f2bf(f[j]);
    }
    const int g = grow / NPG;
#pragma unroll
    for (int i = 0; i < 6; ++i) {              // 48 bf16 = 6 x uint4
      uint4 q = *(const uint4*)(g1b + (size_t)g * 96 + i * 8);
      *(uint4*)&sA[r * 456 + 352 + i * 8] = q;
    }
  }

  f32x8 zero = {};
  f32x8 acc[16];
#pragma unroll
  for (int j = 0; j < 16; ++j) acc[j] = zero;

  const u16* arow = sA + (w * 16 + lrow) * 456;

#if USE_TDM
  if (w == 0) __builtin_amdgcn_s_wait_tensorcnt((short)0);
  __syncthreads();                       // A staged + B slice 0 in LDS
  for (int ks = 0; ks < 14; ++ks) {
    if (w == 0 && ks + 1 < 14) tdm_load_bslice(WT, ks + 1, (ks + 1) & 1);
    const int buf = ks & 1;
    const int kA = ks * 32 + 16 * lhi;
    bf16x16 a = ld_frag2(arow + kA, arow + kA + 8);
    const u16* bbase = sB + buf * (256 * 40) + 16 * lhi;
#pragma unroll
    for (int j = 0; j < 16; ++j) {
      const u16* bp = bbase + (j * 16 + lrow) * 40;
      bf16x16 b = ld_frag2(bp, bp + 8);
      acc[j] = __builtin_amdgcn_wmma_f32_16x16x32_bf16(
          false, a, false, b, (short)0, acc[j], false, false);
    }
    if (w == 0 && ks + 1 < 14) __builtin_amdgcn_s_wait_tensorcnt((short)0);
    __syncthreads();                     // next buffer DMA complete
  }
#else
  // Fallback: cooperative load into regs + ds_store (thread t owns column n=t)
  uint4 breg0, breg1, breg2, breg3;
#define LOADB(ks_) { const uint4* p_ = (const uint4*)(WT + (size_t)t * FCAT + (ks_) * 32); \
                     breg0 = p_[0]; breg1 = p_[1]; breg2 = p_[2]; breg3 = p_[3]; }
#define STOREB(buf_) { uint4* q_ = (uint4*)(sB + ((buf_) * 256 + t) * 40); \
                       q_[0] = breg0; q_[1] = breg1; q_[2] = breg2; q_[3] = breg3; }
  LOADB(0);
  STOREB(0);
  for (int ks = 0; ks < 14; ++ks) {
    if (ks + 1 < 14) LOADB(ks + 1);
    __syncthreads();                     // A + current B buffer visible
    const int buf = ks & 1;
    const int kA = ks * 32 + 16 * lhi;
    bf16x16 a = ld_frag2(arow + kA, arow + kA + 8);
    const u16* bbase = sB + buf * (256 * 40) + 16 * lhi;
#pragma unroll
    for (int j = 0; j < 16; ++j) {
      const u16* bp = bbase + (j * 16 + lrow) * 40;
      bf16x16 b = ld_frag2(bp, bp + 8);
      acc[j] = __builtin_amdgcn_wmma_f32_16x16x32_bf16(
          false, a, false, b, (short)0, acc[j], false, false);
    }
    if (ks + 1 < 14) STOREB((ks + 1) & 1);
  }
#undef LOADB
#undef STOREB
#endif

  // ---- epilogue: bias + relu + store ----
#pragma unroll
  for (int j = 0; j < 16; ++j) {
    int col = j * 16 + lrow;
    float bb = b_out[col];
#pragma unroll
    for (int r = 0; r < 8; ++r) {
      int row = w * 16 + r + 8 * lhi;
      out[(size_t)(row0 + row) * FIN + col] = fmaxf(acc[j][r] + bb, 0.0f);
    }
  }
}

// ---------------------------------------------------------------------------
extern "C" void kernel_launch(void* const* d_in, const int* in_sizes, int n_in,
                              void* d_out, int out_size, void* d_ws, size_t ws_size,
                              hipStream_t stream) {
  const float* x      = (const float*)d_in[0];
  const int*   ei     = (const int*)  d_in[1];
  const float* W1     = (const float*)d_in[4];
  const float* b1     = (const float*)d_in[5];
  const float* W2     = (const float*)d_in[6];
  const float* b2     = (const float*)d_in[7];
  const float* t_tri0 = (const float*)d_in[8];
  const float* t_gau0 = (const float*)d_in[9];
  const float* sig0   = (const float*)d_in[10];
  const float* W_lin0 = (const float*)d_in[11];
  const float* b_lin0 = (const float*)d_in[12];
  const float* c_rh0  = (const float*)d_in[13];
  const float* r_rh0  = (const float*)d_in[14];
  const float* t_tri1 = (const float*)d_in[15];
  const float* t_gau1 = (const float*)d_in[16];
  const float* sig1   = (const float*)d_in[17];
  const float* W_lin1 = (const float*)d_in[18];
  const float* b_lin1 = (const float*)d_in[19];
  const float* c_rh1  = (const float*)d_in[20];
  const float* r_rh1  = (const float*)d_in[21];
  const float* W_out  = (const float*)d_in[22];
  const float* b_out  = (const float*)d_in[23];

  char* ws = (char*)d_ws;
  const size_t OFF_FV = 0;                               // 51200*8*4
  const size_t OFF_XB = OFF_FV + (size_t)NN * KP * 4;    // 51200*256*2
  const size_t OFF_WT = OFF_XB + (size_t)NN * FIN * 2;   // 256*448*2
  const size_t OFF_UP = OFF_WT + (size_t)FIN * FCAT * 2; // 128*8*4
  const size_t OFF_G1 = OFF_UP + (size_t)NG * KP * 4;    // 128*96*2

  float* fv       = (float*)(ws + OFF_FV);
  u16*   xb       = (u16*)  (ws + OFF_XB);
  u16*   WT       = (u16*)  (ws + OFF_WT);
  float* unpaired = (float*)(ws + OFF_UP);
  u16*   g1b      = (u16*)  (ws + OFF_G1);

  wt_kernel<<<FCAT, 256, 0, stream>>>(W_out, WT);
  fv_kernel<<<NN / 128, 256, K0_LDS, stream>>>(x, W1, b1, W2, b2, fv, xb);
  edge_max_kernel<<<NG, 256, 0, stream>>>(ei, fv, unpaired);
  g1_kernel<<<1, 128, 0, stream>>>(unpaired, g1b, t_tri1, t_gau1, sig1,
                                   W_lin1, b_lin1, c_rh1, r_rh1);
  out_gemm_kernel<<<NN / 128, 256, K3_LDS, stream>>>(
      xb, fv, g1b, WT, b_out, t_tri0, t_gau0, sig0, W_lin0, b_lin0, c_rh0,
      r_rh0, (float*)d_out);
}
